// BidirRNNLayer_40063454937397
// MI455X (gfx1250) — 8-bit matrix operand order repaired and checked by reading, compiled, not run on hardware
//
#include <hip/hip_runtime.h>
#include <math.h>


#define RNN_SEQ   2048
#define RNN_BATCH 16
#define RNN_IN    512
#define RNN_HID   512
#define RNN_M     (RNN_SEQ * RNN_BATCH)

typedef __attribute__((ext_vector_type(8))) int v8i;

__device__ __forceinline__ v8i ld_a_global(const signed char* base) {
  int2 t0 = *(const int2*)(base);
  int2 t1 = *(const int2*)(base + 16);
  int2 t2 = *(const int2*)(base + 32);
  int2 t3 = *(const int2*)(base + 48);
  v8i a;
  a[0]=t0.x; a[1]=t0.y; a[2]=t1.x; a[3]=t1.y;
  a[4]=t2.x; a[5]=t2.y; a[6]=t3.x; a[7]=t3.y;
  return a;
}
__device__ __forceinline__ v8i ld_b_global(const signed char* base) {
  int2 t0 = *(const int2*)(base);
  int2 t1 = *(const int2*)(base + 16);
  int2 t2 = *(const int2*)(base + 32);
  int2 t3 = *(const int2*)(base + 48);
  v8i b;
  b[0]=t0.x; b[1]=t0.y; b[2]=t1.x; b[3]=t1.y;
  b[4]=t2.x; b[5]=t2.y; b[6]=t3.x; b[7]=t3.y;
  return b;
}

__global__ void rnn_init_kernel(unsigned* thr) {
  if (threadIdx.x < 2) thr[threadIdx.x] = 0u;
}

__global__ __launch_bounds__(256)
void rnn_absmax_kernel(const float* __restrict__ w_ri_f, const float* __restrict__ w_rh_f,
                       const float* __restrict__ w_ri_b, const float* __restrict__ w_rh_b,
                       unsigned* __restrict__ thr) {
  __shared__ float red[256];
  const int y = blockIdx.y;
  const float* src = (y == 0) ? w_ri_f : (y == 1) ? w_rh_f : (y == 2) ? w_ri_b : w_rh_b;
  const int id = blockIdx.x * 256 + threadIdx.x;
  red[threadIdx.x] = fabsf(src[id]);
  __syncthreads();
  for (int s = 128; s > 0; s >>= 1) {
    if (threadIdx.x < s) red[threadIdx.x] = fmaxf(red[threadIdx.x], red[threadIdx.x + s]);
    __syncthreads();
  }
  if (threadIdx.x == 0)
    atomicMax(&thr[y >> 1], __float_as_uint(red[0]));
}

__global__ __launch_bounds__(256)
void rnn_quant_w_kernel(const float* __restrict__ w_ri_f, const float* __restrict__ w_rh_f,
                        const float* __restrict__ w_ri_b, const float* __restrict__ w_rh_b,
                        const unsigned* __restrict__ thr,
                        signed char* __restrict__ wriT_f, signed char* __restrict__ wrhT_f,
                        signed char* __restrict__ wriT_b, signed char* __restrict__ wrhT_b) {
  const int y = blockIdx.y;
  const float* src = (y == 0) ? w_ri_f : (y == 1) ? w_rh_f : (y == 2) ? w_ri_b : w_rh_b;
  signed char* dst = (y == 0) ? wriT_f : (y == 1) ? wrhT_f : (y == 2) ? wriT_b : wrhT_b;
  const float inv = 127.0f / __uint_as_float(thr[y >> 1]);
  const int id = blockIdx.x * 256 + threadIdx.x;
  const int k = id >> 9, n = id & 511;
  float q = rintf(src[k * 512 + n] * inv);
  q = fminf(fmaxf(q, -127.0f), 127.0f);
  dst[n * 512 + k] = (signed char)(int)q;
}

__global__ __launch_bounds__(256)
void rnn_quant_x_kernel(const float* __restrict__ x, signed char* __restrict__ xi8, int n) {
  const int i = blockIdx.x * 256 + threadIdx.x;
  if (i < n) {
    float v = fminf(fmaxf(x[i], -1.0f), 1.0f);
    float q = rintf(v * 127.0f);
    q = fminf(fmaxf(q, -127.0f), 127.0f);
    xi8[i] = (signed char)(int)q;
  }
}

__global__ __launch_bounds__(256)
void rnn_pregemm_kernel(const signed char* __restrict__ xi8,
                        const signed char* __restrict__ wriT_f,
                        const signed char* __restrict__ wriT_b,
                        int* __restrict__ outI) {
  const int tid  = threadIdx.x;
  const int wave = tid >> 5, lane = tid & 31;
  const int half = lane >> 4, l16 = lane & 15;
  const int tile = blockIdx.y * 8 + wave;
  const int dir  = tile >> 5, nt = tile & 31;
  const signed char* wT = dir ? wriT_b : wriT_f;
  const int n     = nt * 16 + l16;
  const int mbase = blockIdx.x * 16;
  const signed char* arow = xi8 + (size_t)(mbase + l16) * RNN_IN + half * 8;
  const signed char* bcol = wT + (size_t)n * RNN_IN + half * 8;

  v8i acc = {0, 0, 0, 0, 0, 0, 0, 0};
#pragma unroll
  for (int ks = 0; ks < 8; ++ks) {
    v8i a = ld_a_global(arow + ks * 64);
    v8i b = ld_b_global(bcol + ks * 64);
    acc = __builtin_amdgcn_wmma_i32_16x16x64_iu8(true, a, true, b, acc, false, false);
  }
#pragma unroll
  for (int r = 0; r < 8; ++r)
    outI[(size_t)(mbase + r + 8 * half) * (2 * RNN_HID) + dir * RNN_HID + n] = acc[r];
}

__global__ __launch_bounds__(512)
void rnn_recurrence_kernel(const signed char* __restrict__ wrhT_f,
                           const signed char* __restrict__ wrhT_b,
                           const float* __restrict__ b_f,
                           const float* __restrict__ b_b,
                           const unsigned* __restrict__ thr_bits,
                           float* __restrict__ out) {
  __shared__ signed char state[2][RNN_BATCH * RNN_HID];
  const int dir  = blockIdx.x;
  const int tid  = threadIdx.x;
  const int wave = tid >> 5;
  const int lane = tid & 31;
  const int half = lane >> 4;
  const int m    = lane & 15;

  const signed char* wT = dir ? wrhT_b : wrhT_f;
  const float*       bb = dir ? b_b    : b_f;
  const float fac = __uint_as_float(thr_bits[dir]) * (1.0f / (127.0f * 127.0f));

  v8i bfrag[2][8];
  int   ncol[2];
  float bval[2];
#pragma unroll
  for (int j = 0; j < 2; ++j) {
    const int nt = wave * 2 + j;
    ncol[j] = nt * 16 + (lane & 15);
    bval[j] = bb[ncol[j]];
    const signed char* col = wT + (size_t)ncol[j] * RNN_HID + half * 8;
#pragma unroll
    for (int ks = 0; ks < 8; ++ks)
      bfrag[j][ks] = ld_b_global(col + ks * 64);
  }

  int eofs[2][8];
#pragma unroll
  for (int j = 0; j < 2; ++j)
#pragma unroll
    for (int r = 0; r < 8; ++r)
      eofs[j][r] = (r + 8 * half) * (2 * RNN_HID) + dir * RNN_HID + ncol[j];

  for (int i = tid; i < RNN_BATCH * RNN_HID; i += 512) state[0][i] = 0;

  int* outI = (int*)out;

  for (int step = 0; step < RNN_SEQ; ++step) {
    const int t   = dir ? (RNN_SEQ - 1 - step) : step;
    const int cur = step & 1, nxt = cur ^ 1;
    const size_t tbase = (size_t)t * RNN_BATCH * (2 * RNN_HID);
    __syncthreads();

    int pre[2][8];
#pragma unroll
    for (int j = 0; j < 2; ++j)
#pragma unroll
      for (int r = 0; r < 8; ++r)
        pre[j][r] = __builtin_nontemporal_load(&outI[tbase + eofs[j][r]]);

    v8i acc0 = {0, 0, 0, 0, 0, 0, 0, 0};
    v8i acc1 = {0, 0, 0, 0, 0, 0, 0, 0};
#pragma unroll
    for (int ks = 0; ks < 8; ++ks) {
      const signed char* ab = &state[cur][m * RNN_HID + ks * 64 + half * 8];
      int2 t0 = *(const int2*)(ab);
      int2 t1 = *(const int2*)(ab + 16);
      int2 t2 = *(const int2*)(ab + 32);
      int2 t3 = *(const int2*)(ab + 48);
      v8i a;
      a[0]=t0.x; a[1]=t0.y; a[2]=t1.x; a[3]=t1.y;
      a[4]=t2.x; a[5]=t2.y; a[6]=t3.x; a[7]=t3.y;
      acc0 = __builtin_amdgcn_wmma_i32_16x16x64_iu8(true, a, true, bfrag[0][ks], acc0, false, false);
      acc1 = __builtin_amdgcn_wmma_i32_16x16x64_iu8(true, a, true, bfrag[1][ks], acc1, false, false);
    }

#pragma unroll
    for (int j = 0; j < 2; ++j) {
      v8i acc = j ? acc1 : acc0;
#pragma unroll
      for (int r = 0; r < 8; ++r) {
        const int   mm   = r + 8 * half;
        const float gate = (float)(pre[j][r] + acc[r]) * fac + bval[j];
        const float h    = tanhf(gate);
        float q = rintf(fminf(fmaxf(h * 127.0f, -127.0f), 127.0f));
        __builtin_nontemporal_store(q * (1.0f / 127.0f), &out[tbase + eofs[j][r]]);
        state[nxt][mm * RNN_HID + ncol[j]] = (signed char)(int)q;
      }
    }
  }
}

extern "C" void kernel_launch(void* const* d_in, const int* in_sizes, int n_in,
                              void* d_out, int out_size, void* d_ws, size_t ws_size,
                              hipStream_t stream) {
  const float* inputs = (const float*)d_in[0];
  const float* w_ri_f = (const float*)d_in[1];
  const float* w_rh_f = (const float*)d_in[2];
  const float* b_f    = (const float*)d_in[3];
  const float* w_ri_b = (const float*)d_in[4];
  const float* w_rh_b = (const float*)d_in[5];
  const float* b_b    = (const float*)d_in[6];

  char* ws = (char*)d_ws;
  unsigned*    thr    = (unsigned*)ws;
  signed char* xi8    = (signed char*)(ws + 256);
  signed char* wriT_f = xi8 + (size_t)RNN_M * RNN_IN;
  signed char* wrhT_f = wriT_f + (size_t)RNN_IN * RNN_HID;
  signed char* wriT_b = wrhT_f + (size_t)RNN_HID * RNN_HID;
  signed char* wrhT_b = wriT_b + (size_t)RNN_IN * RNN_HID;

  rnn_init_kernel<<<dim3(1), dim3(32), 0, stream>>>(thr);
  rnn_absmax_kernel<<<dim3(1024, 4), dim3(256), 0, stream>>>(
      w_ri_f, w_rh_f, w_ri_b, w_rh_b, thr);
  rnn_quant_w_kernel<<<dim3(1024, 4), dim3(256), 0, stream>>>(
      w_ri_f, w_rh_f, w_ri_b, w_rh_b, thr, wriT_f, wrhT_f, wriT_b, wrhT_b);
  const int nx = RNN_M * RNN_IN;
  rnn_quant_x_kernel<<<dim3((nx + 255) / 256), dim3(256), 0, stream>>>(inputs, xi8, nx);
  rnn_pregemm_kernel<<<dim3(RNN_M / 16, 8), dim3(256), 0, stream>>>(
      xi8, wriT_f, wriT_b, (int*)d_out);
  rnn_recurrence_kernel<<<dim3(2), dim3(512), 0, stream>>>(
      wrhT_f, wrhT_b, b_f, b_b, thr, (float*)d_out);
}
